// GNNModel_33895881900209
// MI455X (gfx1250) — matrix operand order repaired and checked by reading, compiled, not run on hardware
//
#include <hip/hip_runtime.h>
#include <hip/hip_bf16.h>

// GCN 10-layer pipeline for MI455X (gfx1250, wave32).
// - GEMMs: v_wmma_f32_16x16x32_bf16. Per block: 8 waves, LDS-staged weight strip
//   shared by all waves. Per wave: 2 M-tiles x NT N-tiles (8 WMMAs per K-step),
//   B fragments loaded into independent registers so all ds_load_b128 overlap
//   behind a single s_wait_dscnt.
// - Aggregation: CSR built once (graph is layer-invariant), then gather-based
//   fused aggregate+selfloop+bias+skip+relu kernel -- no per-layer f32 atomics.

#define N_NODES 30000
#define N_EDGES 300000

typedef __attribute__((ext_vector_type(16))) __bf16 v16bf;
typedef __attribute__((ext_vector_type(8)))  float  v8f;

union FragAB {
    v16bf v;
    uint4 u[2];
};

// ---------------- graph preprocessing (once per launch) ----------------

__global__ __launch_bounds__(256) void k_zero_int(int* __restrict__ p, int n) {
    int i = blockIdx.x * blockDim.x + threadIdx.x;
    if (i < n) p[i] = 0;
}

__global__ __launch_bounds__(256) void k_hist(const int* __restrict__ dst,
                                              int* __restrict__ cnt, int e) {
    int i = blockIdx.x * blockDim.x + threadIdx.x;
    if (i < e) atomicAdd(&cnt[dst[i]], 1);
}

__global__ __launch_bounds__(256) void k_dis(const int* __restrict__ cnt,
                                             float* __restrict__ dis, int n) {
    int i = blockIdx.x * blockDim.x + threadIdx.x;
    if (i < n) dis[i] = rsqrtf((float)cnt[i] + 1.0f);   // deg + self loop
}

// single-block exclusive scan: rowPtr[0..n-1] = exclusive prefix of cnt, rowPtr[n] = total
__global__ __launch_bounds__(256) void k_scan(const int* __restrict__ cnt,
                                              int* __restrict__ rowPtr, int n, int total) {
    __shared__ int partial[256];
    const int T = 256;
    const int tid = threadIdx.x;
    const int seg = (n + T - 1) / T;
    const int beg = tid * seg;
    const int end = (beg + seg < n) ? (beg + seg) : n;
    int s = 0;
    for (int i = beg; i < end; ++i) s += cnt[i];
    partial[tid] = s;
    __syncthreads();
    if (tid == 0) {
        int run = 0;
        for (int i = 0; i < T; ++i) { int t = partial[i]; partial[i] = run; run += t; }
    }
    __syncthreads();
    int off = partial[tid];
    for (int i = beg; i < end; ++i) { rowPtr[i] = off; off += cnt[i]; }
    if (tid == 0) rowPtr[n] = total;
}

__global__ __launch_bounds__(256) void k_copy_int(const int* __restrict__ a,
                                                  int* __restrict__ b, int n) {
    int i = blockIdx.x * blockDim.x + threadIdx.x;
    if (i < n) b[i] = a[i];
}

__global__ __launch_bounds__(256) void k_fill(const int* __restrict__ src,
                                              const int* __restrict__ dst,
                                              const float* __restrict__ dis,
                                              int* __restrict__ cursor,
                                              int* __restrict__ csrSrc,
                                              float* __restrict__ csrNorm, int e) {
    int i = blockIdx.x * blockDim.x + threadIdx.x;
    if (i >= e) return;
    int s = src[i], d = dst[i];
    int pos = atomicAdd(&cursor[d], 1);
    csrSrc[pos]  = s;
    csrNorm[pos] = dis[s] * dis[d];
}

// ---------------- conversions ----------------

__global__ __launch_bounds__(256) void k_f32_to_bf16(const float* __restrict__ src,
                                                     __hip_bfloat16* __restrict__ dst,
                                                     int n) {
    int i = blockIdx.x * blockDim.x + threadIdx.x;
    if (i < n) dst[i] = __float2bfloat16(src[i]);
}

// W is [K x N] row-major f32; produce Wt [N x K] row-major bf16.
__global__ __launch_bounds__(256) void k_transpose_bf16(const float* __restrict__ W,
                                                        __hip_bfloat16* __restrict__ Wt,
                                                        int K, int N) {
    int idx = blockIdx.x * blockDim.x + threadIdx.x;
    if (idx >= K * N) return;
    int k = idx / N;
    int n = idx - k * N;
    Wt[(size_t)n * K + k] = __float2bfloat16(W[idx]);
}

// ---------------- WMMA GEMM: H[M x N] = A[M x K](bf16) * Wt[N x K]^T(bf16) ----------------
// Block: 8 wave32s, 256 threads. Each wave: MT=2 M-tiles (16 rows each), NT N-tiles.
// Weight strip (NT*16 rows of Wt x 64 K) staged in LDS, shared by all 8 waves.
// Fragment layouts per CDNA5 ISA 7.12.2 (A: lane=row, lane[4]=K-half; D: lane=col).
template <int NT>
__global__ __launch_bounds__(256) void k_gemm_wmma(const __hip_bfloat16* __restrict__ A,
                                                   const __hip_bfloat16* __restrict__ Bt,
                                                   float* __restrict__ H,
                                                   int M, int K, int N) {
    constexpr int MT    = 2;
    constexpr int NROWS = NT * 16;
    constexpr int ROWB  = 64 * 2 + 16;   // 64 bf16 + 16B pad -> conflict-free ds_load_b128
    __shared__ __align__(16) unsigned char ldsB[NROWS * ROWB];

    const int lane   = threadIdx.x & 31;
    const int wave   = threadIdx.x >> 5;
    const int tilesM = M >> 4;

    int tM[MT];
#pragma unroll
    for (int mt = 0; mt < MT; ++mt) {
        int t = blockIdx.y * (8 * MT) + wave * MT + mt;
        tM[mt] = (t < tilesM) ? t : (tilesM - 1);  // uniform clamp: duplicates recompute
    }                                              // identical values; barriers stay uniform
    const int nBase = blockIdx.x * NROWS;

    const int r  = lane & 15;
    const int kh = lane >> 4;

    const __bf16* arow[MT];
#pragma unroll
    for (int mt = 0; mt < MT; ++mt)
        arow[mt] = (const __bf16*)(A + (size_t)(tM[mt] * 16 + r) * K);

    v8f acc[MT][NT];
    v8f zero = {};
#pragma unroll
    for (int mt = 0; mt < MT; ++mt)
#pragma unroll
        for (int t = 0; t < NT; ++t) acc[mt][t] = zero;

    for (int k0 = 0; k0 < K; k0 += 64) {
        const int kc  = (K - k0 < 64) ? (K - k0) : 64;   // 64 except K==32 case
        const int prs = (kc == 64) ? 3 : 2;              // log2(16B chunks per row)
        const int total = NROWS << prs;
        for (int ch = threadIdx.x; ch < total; ch += 256) {
            const int row = ch >> prs;
            const int kb  = ch & ((1 << prs) - 1);
            const uint4 v = *(const uint4*)(Bt + (size_t)(nBase + row) * K + k0 + kb * 8);
            *(uint4*)(ldsB + row * ROWB + kb * 16) = v;
        }
        __syncthreads();

        if (k0 + 64 < K) {
#pragma unroll
            for (int mt = 0; mt < MT; ++mt)
                __builtin_prefetch(arow[mt] + k0 + 64, 0, 3);   // global_prefetch_b8
        }

#pragma unroll
        for (int kk = 0; kk < 64; kk += 32) {
            if (kk < kc) {
                // A fragments (global, L2-resident activations)
                FragAB a[MT];
#pragma unroll
                for (int mt = 0; mt < MT; ++mt) {
                    a[mt].u[0] = *(const uint4*)(arow[mt] + k0 + kk + kh * 8);
                    a[mt].u[1] = *(const uint4*)(arow[mt] + k0 + kk + 16 + kh * 8);
                }
                // B fragments into independent registers: all ds_load_b128 overlap
                FragAB b[NT];
#pragma unroll
                for (int t = 0; t < NT; ++t) {
                    const unsigned char* lrow =
                        ldsB + (t * 16 + r) * ROWB + (kk + kh * 8) * 2;
                    b[t].u[0] = *(const uint4*)(lrow);
                    b[t].u[1] = *(const uint4*)(lrow + 32);
                }
#pragma unroll
                for (int mt = 0; mt < MT; ++mt)
#pragma unroll
                    for (int t = 0; t < NT; ++t)
                        acc[mt][t] = __builtin_amdgcn_wmma_f32_16x16x32_bf16(
                            false, a[mt].v, false, b[t].v, (short)0, acc[mt][t],
                            false, false);
            }
        }
        __syncthreads();
    }

#pragma unroll
    for (int mt = 0; mt < MT; ++mt) {
        const int mBase = tM[mt] * 16 + kh * 8;
#pragma unroll
        for (int t = 0; t < NT; ++t) {
            const int n = nBase + t * 16 + r;
            float* out = H + (size_t)mBase * N + n;
#pragma unroll
            for (int rr = 0; rr < 8; ++rr)
                out[(size_t)rr * N] = acc[mt][t][rr];
        }
    }
}

// ---------------- fused gather-aggregate + self-loop + bias + skip + relu ----------------
// One thread per (node, 4 channels). H rows of neighbors are L2-resident (<=123MB < 192MB).
__global__ __launch_bounds__(256) void k_agg_finalize(const int* __restrict__ rowPtr,
                                                      const int* __restrict__ csrSrc,
                                                      const float* __restrict__ csrNorm,
                                                      const float* __restrict__ H,
                                                      const float* __restrict__ dis,
                                                      const float* __restrict__ bias,
                                                      const __hip_bfloat16* __restrict__ skip,
                                                      __hip_bfloat16* __restrict__ outBf,
                                                      float* __restrict__ outF,
                                                      int Nn, int C, int cshift) {
    size_t idx = (size_t)blockIdx.x * blockDim.x + threadIdx.x;
    size_t total = (size_t)Nn << (cshift - 2);
    if (idx >= total) return;
    const int d = (int)(idx >> (cshift - 2));
    const int c = (int)(idx & ((size_t)(C >> 2) - 1)) << 2;

    float sx = 0.f, sy = 0.f, sz = 0.f, sw = 0.f;
    const int beg = rowPtr[d], end = rowPtr[d + 1];
    for (int j = beg; j < end; ++j) {
        const int   s  = csrSrc[j];
        const float nm = csrNorm[j];
        const float4 hv = *(const float4*)(H + ((size_t)s << cshift) + c);
        sx += hv.x * nm; sy += hv.y * nm; sz += hv.z * nm; sw += hv.w * nm;
    }
    const size_t o  = ((size_t)d << cshift) + c;
    const float  di = dis[d];
    const float  sc = di * di;
    const float4 hd = *(const float4*)(H + o);
    const float4 bv = *(const float4*)(bias + c);
    float vx = sx + hd.x * sc + bv.x;
    float vy = sy + hd.y * sc + bv.y;
    float vz = sz + hd.z * sc + bv.z;
    float vw = sw + hd.w * sc + bv.w;
    if (skip) {
        vx += __bfloat162float(skip[o + 0]);
        vy += __bfloat162float(skip[o + 1]);
        vz += __bfloat162float(skip[o + 2]);
        vw += __bfloat162float(skip[o + 3]);
    }
    vx = fmaxf(vx, 0.f); vy = fmaxf(vy, 0.f); vz = fmaxf(vz, 0.f); vw = fmaxf(vw, 0.f);
    if (outBf) {
        outBf[o + 0] = __float2bfloat16(vx);
        outBf[o + 1] = __float2bfloat16(vy);
        outBf[o + 2] = __float2bfloat16(vz);
        outBf[o + 3] = __float2bfloat16(vw);
    }
    if (outF) {
        outF[o + 0] = vx; outF[o + 1] = vy; outF[o + 2] = vz; outF[o + 3] = vw;
    }
}

// ---------------- host ----------------

extern "C" void kernel_launch(void* const* d_in, const int* in_sizes, int n_in,
                              void* d_out, int out_size, void* d_ws, size_t ws_size,
                              hipStream_t stream) {
    (void)in_sizes; (void)n_in; (void)out_size; (void)ws_size;
    const int Nn = N_NODES, E = N_EDGES;
    const int CH[11] = {32, 1024, 512, 256, 128, 64, 128, 256, 512, 1024, 32};

    const float* x    = (const float*)d_in[0];
    const int*   edge = (const int*)d_in[1];
    const int*   esrc = edge;
    const int*   edst = edge + E;
    const float* W[10];
    const float* bvec[10];
    for (int i = 0; i < 10; ++i) {
        W[i]    = (const float*)d_in[2 + i];
        bvec[i] = (const float*)d_in[12 + i];
    }

    char*  ws  = (char*)d_ws;
    size_t off = 0;
    auto alloc = [&](size_t bytes) -> void* {
        void* p = ws + off;
        off += (bytes + 255) & ~(size_t)255;
        return p;
    };

    float* dis     = (float*)alloc((size_t)Nn * 4);
    int*   cnt     = (int*)alloc((size_t)Nn * 4);
    int*   rowPtr  = (int*)alloc((size_t)(Nn + 1) * 4);
    int*   cursor  = (int*)alloc((size_t)Nn * 4);
    int*   csrSrc  = (int*)alloc((size_t)E * 4);
    float* csrNorm = (float*)alloc((size_t)E * 4);

    __hip_bfloat16* x0b = (__hip_bfloat16*)alloc((size_t)Nn * 32 * 2);
    __hip_bfloat16* Wt[10];
    for (int i = 0; i < 10; ++i)
        Wt[i] = (__hip_bfloat16*)alloc((size_t)CH[i] * CH[i + 1] * 2);
    float*          Hbuf = (float*)alloc((size_t)Nn * 1024 * 4);
    __hip_bfloat16* X1   = (__hip_bfloat16*)alloc((size_t)Nn * 1024 * 2);
    __hip_bfloat16* X2   = (__hip_bfloat16*)alloc((size_t)Nn * 512 * 2);
    __hip_bfloat16* X3   = (__hip_bfloat16*)alloc((size_t)Nn * 256 * 2);
    __hip_bfloat16* X4   = (__hip_bfloat16*)alloc((size_t)Nn * 128 * 2);
    __hip_bfloat16* AA   = (__hip_bfloat16*)alloc((size_t)Nn * 1024 * 2);
    __hip_bfloat16* AB   = (__hip_bfloat16*)alloc((size_t)Nn * 512 * 2);

    const int T = 256;

    // ---- CSR build + D^{-1/2} (graph is identical for all 10 layers) ----
    k_zero_int<<<(Nn + T - 1) / T, T, 0, stream>>>(cnt, Nn);
    k_hist    <<<(E + T - 1) / T, T, 0, stream>>>(edst, cnt, E);
    k_dis     <<<(Nn + T - 1) / T, T, 0, stream>>>(cnt, dis, Nn);
    k_scan    <<<1, T, 0, stream>>>(cnt, rowPtr, Nn, E);
    k_copy_int<<<(Nn + T - 1) / T, T, 0, stream>>>(rowPtr, cursor, Nn);
    k_fill    <<<(E + T - 1) / T, T, 0, stream>>>(esrc, edst, dis, cursor, csrSrc, csrNorm, E);

    // ---- input -> bf16, weights -> transposed bf16 ----
    k_f32_to_bf16<<<(Nn * 32 + T - 1) / T, T, 0, stream>>>(x, x0b, Nn * 32);
    for (int i = 0; i < 10; ++i) {
        int nel = CH[i] * CH[i + 1];
        k_transpose_bf16<<<(nel + T - 1) / T, T, 0, stream>>>(W[i], Wt[i], CH[i], CH[i + 1]);
    }

    const __hip_bfloat16* inb[10]  = {x0b, X1, X2, X3, X4, AA, AB, AA, AB, AA};
    __hip_bfloat16*       outb[10] = {X1,  X2, X3, X4, AA, AB, AA, AB, AA, nullptr};
    const __hip_bfloat16* skip[10] = {nullptr, nullptr, nullptr, nullptr, nullptr,
                                      X4, X3, X2, X1, nullptr};

    const int gy = (Nn / 16 + 15) / 16;  // 118 blocks of 16 M-tiles (8 waves x MT=2)

    for (int l = 0; l < 10; ++l) {
        const int K = CH[l], C = CH[l + 1];
        const int cshift = __builtin_ctz(C);

        // GEMM (WMMA + LDS-staged weights): Hbuf[Nn x C] = inb * Wt^T
        if (C >= 64) {
            dim3 grid(C / 64, gy);
            k_gemm_wmma<4><<<grid, 256, 0, stream>>>(inb[l], Wt[l], Hbuf, Nn, K, C);
        } else {
            dim3 grid(C / 32, gy);
            k_gemm_wmma<2><<<grid, 256, 0, stream>>>(inb[l], Wt[l], Hbuf, Nn, K, C);
        }

        // fused gather-aggregate + self-loop + bias + skip + relu
        size_t nw = (size_t)Nn * (C / 4);
        float* outF = (l == 9) ? (float*)d_out : nullptr;
        k_agg_finalize<<<(unsigned)((nw + T - 1) / T), T, 0, stream>>>(
            rowPtr, csrSrc, csrNorm, Hbuf, dis, bvec[l], skip[l], outb[l], outF,
            Nn, C, cshift);
    }
}
